// CustomGRU_18554258718729
// MI455X (gfx1250) — matrix operand order repaired and checked by reading, not compiled, not run on hardware
//
#include <hip/hip_runtime.h>
#include <hip/hip_bf16.h>

// ---------------------------------------------------------------------------
// CDNA5 (gfx1250) GRU: bf16 WMMA GEMMs + persistent-scan kernel.
// ~107 GF of GEMM work; weights (~1.6MB bf16) are L2-resident; the 512-step
// recurrence is the serial bottleneck -> one persistent kernel, 4 software
// grid barriers per step. A-operand slabs staged in LDS by the Tensor Data
// Mover (tensor_load_to_lds + s_wait_tensorcnt), freeing the VMEM pipe in the
// latency-critical loop.
// ---------------------------------------------------------------------------

typedef __attribute__((ext_vector_type(16))) __bf16 v16bf;
typedef __attribute__((ext_vector_type(8)))  __bf16 v8bf;
typedef __attribute__((ext_vector_type(8)))  float  v8f;

typedef __attribute__((ext_vector_type(4))) unsigned int tdm_u4;
typedef __attribute__((ext_vector_type(8))) int          tdm_i8;
typedef __attribute__((ext_vector_type(4))) int          tdm_i4;

constexpr int kB = 128, kS = 512, kI = 128, kH = 256, kL = 2;
constexpr int K2H = 2 * kH;               // 512

constexpr int SCAN_BLOCKS = 16;
constexpr int SCAN_TPB    = 512;          // 16 wave32 per block -> 256 waves total

// ---- workspace layout (bytes, 256-aligned) --------------------------------
constexpr size_t oWinT = 0;                                     // 256*128 bf16
constexpr size_t oWzrT = oWinT + (size_t)256 * 128 * 2;         // 2*512*512 bf16
constexpr size_t oWnT  = oWzrT + (size_t)2 * 512 * 512 * 2;     // 2*256*512 bf16
constexpr size_t oXp   = oWnT  + (size_t)2 * 256 * 512 * 2;     // B*S*H bf16
constexpr size_t oHst  = oXp   + (size_t)kB * kS * kH * 2;      // 2L*2par*B*H bf16
constexpr size_t oZr   = oHst  + (size_t)kL * 2 * kB * kH * 2;  // L*B*2H f32
constexpr size_t oSync = oZr   + (size_t)kL * kB * K2H * 4;     // 2 u32

// ---------------------------------------------------------------------------
// WMMA fragment packing (ISA 7.12.2, wave32):
//  A 16x32 bf16: lane holds row M=lane%16; elems[0..7]=K[c*8 .. +7],
//                elems[8..15]=K[c*8+16 ..], c = lane/16.
//  B 32x16 bf16 (from W^T row-major [N][K]): lane holds col N=lane%16;
//                elems[0..7]=K[c*8 .. +7], elems[8..15]=K[c*8+16 ..], c = lane/16.
//  C/D 16x16 f32: elem v at (M = v + 8*(lane/16), N = lane%16).
// ---------------------------------------------------------------------------
static __device__ inline v16bf pack_a_bf16(const __bf16* rowk, int lane) {
  const int o = (lane >> 4) << 3;
  v8bf lo = *(const v8bf*)(rowk + o);
  v8bf hi = *(const v8bf*)(rowk + o + 16);
  v16bf a;
#pragma unroll
  for (int i = 0; i < 8; ++i) { a[i] = lo[i]; a[i + 8] = hi[i]; }
  return a;
}

static __device__ inline v16bf pack_a_f32(const float* rowk, int lane) {
  const int o = (lane >> 4) << 3;
  v16bf a;
#pragma unroll
  for (int i = 0; i < 8; ++i) {
    a[i]     = (__bf16)rowk[o + i];
    a[i + 8] = (__bf16)rowk[o + 16 + i];
  }
  return a;
}

static __device__ inline v16bf pack_b_bf16(const __bf16* rowk, int lane) {
  const int o = (lane >> 4) << 3;
  v8bf lo = *(const v8bf*)(rowk + o);
  v8bf hi = *(const v8bf*)(rowk + o + 16);
  v16bf b;
#pragma unroll
  for (int i = 0; i < 8; ++i) { b[i] = lo[i]; b[i + 8] = hi[i]; }
  return b;
}

static __device__ inline v8f wmma_bf16(v16bf a, v16bf b, v8f c) {
  return __builtin_amdgcn_wmma_f32_16x16x32_bf16(false, a, false, b, (short)0, c,
                                                 false, false);
}

// ---------------------------------------------------------------------------
// TDM: async 2-D tile load global->LDS (ISA ch.8 D# descriptor, groups 0/1;
// groups 2/3 zero => 2-D tensor). data_size=1 (2-byte elems), no padding,
// no iteration, no multicast. Issued by one wave; tracked by TENSORcnt.
// This toolchain declares the 6-arg builtin:
//   (uint32x4 g0, int32x8 g1, int32x4, int32x4, int32x8, i32 cpol)
// ---------------------------------------------------------------------------
static __device__ inline void tdm_load_2d(unsigned lds_byte, const void* gaddr,
                                          unsigned tile_w /*elems*/, unsigned tile_h,
                                          unsigned long long stride_elems) {
  const unsigned long long ga = (unsigned long long)gaddr;
  tdm_u4 g0;
  g0[0] = 1u;                                             // count=1 (valid user D#)
  g0[1] = lds_byte;                                       // lds_addr
  g0[2] = (unsigned)(ga & 0xFFFFFFFFu);                   // global_addr[31:0]
  g0[3] = (unsigned)((ga >> 32) & 0x01FFFFFFu)            // global_addr[56:32]
        | (2u << 30);                                     // type=2 ("image")
  tdm_i8 g1;
  g1[0] = (int)(1u << 16);                                // data_size=1 (2B)
  g1[1] = (int)((tile_w & 0xFFFFu) << 16);                // tensor_dim0[15:0]
  g1[2] = (int)(((tile_w >> 16) & 0xFFFFu)                // tensor_dim0[31:16]
        | ((tile_h & 0xFFFFu) << 16));                    // tensor_dim1[15:0]
  g1[3] = (int)(((tile_h >> 16) & 0xFFFFu)                // tensor_dim1[31:16]
        | ((tile_w & 0xFFFFu) << 16));                    // tile_dim0
  g1[4] = (int)(tile_h & 0xFFFFu);                        // tile_dim1 (tile_dim2=0)
  g1[5] = (int)(stride_elems & 0xFFFFFFFFu);              // dim0_stride[31:0]
  g1[6] = (int)((stride_elems >> 32) & 0xFFFFu);          // dim0_stride[47:32]
  g1[7] = 0;
  tdm_i4 z4 = {0, 0, 0, 0};
  tdm_i8 z8 = {0, 0, 0, 0, 0, 0, 0, 0};
  __builtin_amdgcn_tensor_load_to_lds(g0, g1, z4, z4, z8, 0);
}

static __device__ inline unsigned lds_offset(const void* p) {
  return (unsigned)(size_t)p;                             // flat LDS: addr[31:0]
}

// ---- software grid barrier (SCAN_BLOCKS arrivals, generation counter) -----
static __device__ inline void grid_barrier(unsigned* cnt, volatile unsigned* gen) {
  __threadfence();          // release this thread's stores
  __syncthreads();
  if (threadIdx.x == 0) {
    unsigned g = *gen;
    if (atomicAdd(cnt, 1u) == (unsigned)(SCAN_BLOCKS - 1)) {
      *cnt = 0;
      __threadfence();
      atomicAdd((unsigned*)gen, 1u);
    } else {
      while (*gen == g) { __builtin_amdgcn_s_sleep(1); }
    }
    __threadfence();        // acquire
  }
  __syncthreads();
}

// ---------------------------------------------------------------------------
__global__ void gru_init(__bf16* h, unsigned* sync) {
  const int tid = blockIdx.x * blockDim.x + threadIdx.x;
  const int str = gridDim.x * blockDim.x;
  for (int i = tid; i < kL * 2 * kB * kH; i += str) h[i] = (__bf16)0.0f;
  if (tid == 0) { sync[0] = 0u; sync[1] = 0u; }
}

// Transpose + bf16-convert weights into [N][K] row-major, z|r fused on N.
__global__ void gru_prep_weights(const float* W_in, const float* Wz, const float* Wr,
                                 const float* Wn, __bf16* WinT, __bf16* WzrT,
                                 __bf16* WnT) {
  const int tid = blockIdx.x * blockDim.x + threadIdx.x;
  const int str = gridDim.x * blockDim.x;
  for (int i = tid; i < 256 * 128; i += str) {          // WinT[n][k], W_in[k][n]
    int n = i >> 7, k = i & 127;
    WinT[i] = (__bf16)W_in[k * kH + n];
  }
  for (int i = tid; i < 2 * 512 * 512; i += str) {      // WzrT[l][n][k]
    int l = i >> 18, r = i & ((1 << 18) - 1), n = r >> 9, k = r & 511;
    float v = (n < kH) ? Wz[(size_t)l * K2H * kH + (size_t)k * kH + n]
                       : Wr[(size_t)l * K2H * kH + (size_t)k * kH + (n - kH)];
    WzrT[i] = (__bf16)v;
  }
  for (int i = tid; i < 2 * 256 * 512; i += str) {      // WnT[l][n][k]
    int l = i >> 17, n = (i >> 9) & 255, k = i & 511;
    WnT[i] = (__bf16)Wn[(size_t)l * K2H * kH + (size_t)k * kH + n];
  }
}

// xp = x @ W_in + b_in : [B*S=65536, K=128] x [128, 256] -> bf16
// Each wave: one A fragment shared across 4 N-tiles (4 independent acc chains).
__global__ void gru_xp_gemm(const float* __restrict__ x, const __bf16* __restrict__ WinT,
                            const float* __restrict__ b_in, __bf16* __restrict__ xp) {
  const int lane   = threadIdx.x & 31;
  const int gwave  = (blockIdx.x * blockDim.x + threadIdx.x) >> 5;
  const int nwaves = (gridDim.x * blockDim.x) >> 5;
  const int ntasks = (kB * kS / 16) * 4;                // 4096 tm x 4 tn-groups
  for (int task = gwave; task < ntasks; task += nwaves) {
    const int tm  = task >> 2, tng = task & 3;
    const int mrow = tm * 16 + (lane & 15);
    const int n0   = tng * 64 + (lane & 15);
    const float* arow = x + (size_t)mrow * kI;
    v8f acc[4] = {{}, {}, {}, {}};
#pragma unroll
    for (int kk = 0; kk < kI / 32; ++kk) {
      v16bf a = pack_a_f32(arow + kk * 32, lane);
#pragma unroll
      for (int j = 0; j < 4; ++j) {
        v16bf b = pack_b_bf16(WinT + (size_t)(n0 + j * 16) * kI + kk * 32, lane);
        acc[j] = wmma_bf16(a, b, acc[j]);
      }
    }
#pragma unroll
    for (int j = 0; j < 4; ++j) {
      const int n = n0 + j * 16;
      const float bias = b_in[n];
#pragma unroll
      for (int v = 0; v < 8; ++v) {
        int m = tm * 16 + v + ((lane >> 4) << 3);
        xp[(size_t)m * kH + n] = (__bf16)(acc[j][v] + bias);
      }
    }
  }
}

// Persistent scan: 512 steps x 2 layers x (phaseA zr-GEMM, phaseB n-GEMM+update)
// LDS slab = two contiguous 16x256 bf16 halves: [0]=inp rows, [8KB]=h / r*h.
// Inp/h halves staged by TDM (tensor_load_to_lds), r*h half computed in-place.
__global__ void __launch_bounds__(SCAN_TPB, 1)
gru_scan(const __bf16* __restrict__ xp, const __bf16* __restrict__ WzrT,
         const __bf16* __restrict__ WnT, const float* __restrict__ bz,
         const float* __restrict__ br, const float* __restrict__ bn,
         __bf16* __restrict__ h, float* __restrict__ zr, float* __restrict__ out,
         unsigned* sync) {
  __shared__ __bf16 slab[2 * 16 * kH];                  // 16KB: two 16x256 halves
  const int lane  = threadIdx.x & 31;
  const int gwave = blockIdx.x * (SCAN_TPB >> 5) + (threadIdx.x >> 5);   // 0..255
  const unsigned ldsX = lds_offset(&slab[0]);
  const unsigned ldsH = lds_offset(&slab[16 * kH]);
  unsigned* cnt = sync;
  volatile unsigned* gen = sync + 1;

  for (int t = 0; t < kS; ++t) {
    const int p = t & 1;
    for (int l = 0; l < kL; ++l) {
      const __bf16* hp = h + (size_t)(l * 2 + p) * kB * kH;        // h_prev
      __bf16*       hn = h + (size_t)(l * 2 + (1 - p)) * kB * kH;  // h_new
      const __bf16* inp;
      size_t istride;
      if (l == 0) { inp = xp + (size_t)t * kH; istride = (size_t)kS * kH; }
      else        { inp = h + (size_t)(0 * 2 + (1 - p)) * kB * kH; istride = kH; }
      float* zrl = zr + (size_t)l * kB * K2H;

      // -------- phase A: [z|r] = sigmoid([inp|h_prev] @ WzrT + [bz|br]) ----
      {
        const int m0 = (blockIdx.x >> 1) * 16;          // block's A slab rows
        if (threadIdx.x < 32) {                         // wave 0 drives the TDM
          tdm_load_2d(ldsX, inp + (size_t)m0 * istride, kH, 16, istride);
          tdm_load_2d(ldsH, hp + (size_t)m0 * kH, kH, 16, (unsigned long long)kH);
          __builtin_amdgcn_s_wait_tensorcnt(0);
        }
        __syncthreads();

        const int tm = gwave >> 5, tn = gwave & 31;     // 8 x 32 tiles
        const int n  = tn * 16 + (lane & 15);
        const __bf16* brow  = WzrT + (size_t)l * K2H * K2H + (size_t)n * K2H;
        const __bf16* aldsX = &slab[(lane & 15) * kH];
        const __bf16* aldsH = &slab[16 * kH + (lane & 15) * kH];
        v8f acc = {};
#pragma unroll
        for (int kk = 0; kk < 16; ++kk) {               // kk const after unroll
          const __bf16* arow = (kk < 8) ? aldsX + kk * 32 : aldsH + (kk - 8) * 32;
          v16bf a = pack_a_bf16(arow, lane);
          v16bf b = pack_b_bf16(brow + kk * 32, lane);
          acc = wmma_bf16(a, b, acc);
        }
        const float bias = (n < kH) ? bz[l * kH + n] : br[l * kH + (n - kH)];
#pragma unroll
        for (int v = 0; v < 8; ++v) {
          int m = tm * 16 + v + ((lane >> 4) << 3);
          zrl[(size_t)m * K2H + n] = 1.0f / (1.0f + expf(-(acc[v] + bias)));
        }
      }
      grid_barrier(cnt, gen);

      // -------- phase B: n = tanh([inp | r*h_prev] @ WnT + bn); update h ---
      if (blockIdx.x < 8) {
        const int m0 = blockIdx.x * 16;
        if (threadIdx.x < 32) {                         // inp half via TDM
          tdm_load_2d(ldsX, inp + (size_t)m0 * istride, kH, 16, istride);
          __builtin_amdgcn_s_wait_tensorcnt(0);
        }
        // r*h_prev half: one 8-elem chunk per thread (16 rows x 32 chunks)
        {
          const int row = threadIdx.x >> 5;
          const int kc  = (threadIdx.x & 31) << 3;
          const __bf16* hsrc = hp + (size_t)(m0 + row) * kH + kc;
          const float*  rsrc = zrl + (size_t)(m0 + row) * K2H + kH + kc;
          v8bf d;
#pragma unroll
          for (int i = 0; i < 8; ++i) d[i] = (__bf16)((float)hsrc[i] * rsrc[i]);
          *(v8bf*)&slab[16 * kH + row * kH + kc] = d;
        }
        __syncthreads();

        const int tn = (threadIdx.x >> 5);              // wave id 0..15 = N tile
        const int n  = tn * 16 + (lane & 15);
        const __bf16* brow  = WnT + (size_t)l * kH * K2H + (size_t)n * K2H;
        const __bf16* aldsX = &slab[(lane & 15) * kH];
        const __bf16* aldsH = &slab[16 * kH + (lane & 15) * kH];
        v8f acc = {};
#pragma unroll
        for (int kk = 0; kk < 16; ++kk) {
          const __bf16* arow = (kk < 8) ? aldsX + kk * 32 : aldsH + (kk - 8) * 32;
          v16bf a = pack_a_bf16(arow, lane);
          v16bf b = pack_b_bf16(brow + kk * 32, lane);
          acc = wmma_bf16(a, b, acc);
        }
        const float bias = bn[l * kH + n];
#pragma unroll
        for (int v = 0; v < 8; ++v) {
          int m = m0 + v + ((lane >> 4) << 3);
          float nval  = tanhf(acc[v] + bias);
          float z     = zrl[(size_t)m * K2H + n];
          float hprev = (float)hp[(size_t)m * kH + n];
          float hnew  = (1.0f - z) * nval + z * hprev;
          hn[(size_t)m * kH + n] = (__bf16)hnew;
          if (l == 1) out[(size_t)m * kS * kH + (size_t)t * kH + n] = hnew;
        }
      }
      grid_barrier(cnt, gen);
    }
  }

  // final hidden states -> d_out tail [L,B,H] fp32 (last write parity == 0)
  const int tid = blockIdx.x * blockDim.x + threadIdx.x;
  const int str = gridDim.x * blockDim.x;
  for (int i = tid; i < kL * kB * kH; i += str) {
    int l = i / (kB * kH), rem = i - l * (kB * kH);
    out[(size_t)kB * kS * kH + i] = (float)h[(size_t)(l * 2 + 0) * kB * kH + rem];
  }
}

// ---------------------------------------------------------------------------
extern "C" void kernel_launch(void* const* d_in, const int* in_sizes, int n_in,
                              void* d_out, int out_size, void* d_ws, size_t ws_size,
                              hipStream_t stream) {
  (void)in_sizes; (void)n_in; (void)out_size; (void)ws_size;
  const float* x    = (const float*)d_in[0];
  const float* W_in = (const float*)d_in[1];
  const float* b_in = (const float*)d_in[2];
  const float* Wz   = (const float*)d_in[3];
  const float* bz   = (const float*)d_in[4];
  const float* Wr   = (const float*)d_in[5];
  const float* br   = (const float*)d_in[6];
  const float* Wn   = (const float*)d_in[7];
  const float* bn   = (const float*)d_in[8];
  float* out = (float*)d_out;

  char* ws = (char*)d_ws;
  __bf16*   WinT = (__bf16*)(ws + oWinT);
  __bf16*   WzrT = (__bf16*)(ws + oWzrT);
  __bf16*   WnT  = (__bf16*)(ws + oWnT);
  __bf16*   xp   = (__bf16*)(ws + oXp);
  __bf16*   hbuf = (__bf16*)(ws + oHst);
  float*    zrb  = (float*)(ws + oZr);
  unsigned* sync = (unsigned*)(ws + oSync);

  gru_init<<<64, 256, 0, stream>>>(hbuf, sync);
  gru_prep_weights<<<256, 256, 0, stream>>>(W_in, Wz, Wr, Wn, WinT, WzrT, WnT);
  gru_xp_gemm<<<1024, 256, 0, stream>>>(x, WinT, b_in, xp);
  gru_scan<<<SCAN_BLOCKS, SCAN_TPB, 0, stream>>>(xp, WzrT, WnT, bz, br, bn, hbuf,
                                                 zrb, out, sync);
}
